// cross_attention_lstm_correct_efficient_83141976916059
// MI455X (gfx1250) — matrix operand order repaired and checked by reading, not compiled, not run on hardware
//
#include <hip/hip_runtime.h>
#include <math.h>

#define BB   128
#define DD   1280
#define KK   49
#define VV   10000
#define TT   40
#define VPAD 10240

typedef __bf16 bf16_t;
typedef __attribute__((ext_vector_type(8)))  bf16_t v8bf;
typedef __attribute__((ext_vector_type(16))) bf16_t v16bf;
typedef __attribute__((ext_vector_type(8)))  float  v8f;

static __device__ __forceinline__ float bf2f(bf16_t b) {
  unsigned short s = __builtin_bit_cast(unsigned short, b);
  unsigned u = ((unsigned)s) << 16;
  return __builtin_bit_cast(float, u);
}
static __device__ __forceinline__ bf16_t f2bf(float f) {
  unsigned u = __builtin_bit_cast(unsigned, f);
  unsigned r = u + 0x7FFFu + ((u >> 16) & 1u);
  unsigned short s = (unsigned short)(r >> 16);
  return __builtin_bit_cast(bf16_t, s);
}
static __device__ __forceinline__ float sigm(float x) { return 1.0f / (1.0f + expf(-x)); }

static __device__ __forceinline__ v16bf cat16(v8bf lo, v8bf hi) {
  return __builtin_shufflevector(lo, hi, 0,1,2,3,4,5,6,7,8,9,10,11,12,13,14,15);
}
// A fragment (ISA 16-bit A layout): lane m=lane&15, K = {kh..kh+7, kh+16..kh+23}
static __device__ __forceinline__ v16bf ldA(const bf16_t* p) {
  return cat16(*(const v8bf*)p, *(const v8bf*)(p + 16));
}
// B fragment (ISA 16-bit B layout): lane n=lane&15, K = {kb..kb+7, kb+16..kb+23}
static __device__ __forceinline__ v16bf ldB(const bf16_t* p) {
  return cat16(*(const v8bf*)p, *(const v8bf*)(p + 16));
}
static __device__ __forceinline__ v8f wmma_bf16(v16bf a, v16bf b, v8f c) {
  return __builtin_amdgcn_wmma_f32_16x16x32_bf16(false, a, false, b, (short)0, c, false, false);
}

// ---------------------------------------------------------------------------
// bf16 WMMA GEMM: C[M,N] = A[M,K] * B^T (B stored transposed [N][K]).
// Wave tile 16x64 (4 v8f accs). Block = 4 waves. grid = (Ntiles64/4, M/16).
// K-loop is software-pipelined with ping-pong fragment buffers so global
// loads for chunk i+1 overlap the WMMAs of chunk i (partial loadcnt waits).
// ---------------------------------------------------------------------------
enum { EPI_GATES = 0, EPI_LOGITS = 1, EPI_QTANH = 2, EPI_GATE2 = 3 };

template <int EPI>
__global__ __launch_bounds__(128) void gemm_wmma(
    const bf16_t* __restrict__ A, int lda,
    const bf16_t* __restrict__ B0, const bf16_t* __restrict__ B1,
    int Kdim, int Npad,
    float* __restrict__ outF, bf16_t* __restrict__ outB,
    const float* __restrict__ bias0, const float* __restrict__ bias1,
    int tcol)
{
  const int lane  = threadIdx.x & 31;
  const int wv    = threadIdx.x >> 5;
  const int mtile = blockIdx.y;
  const int ntile = blockIdx.x * 4 + wv;
  const int nbase = ntile * 64;

  const bf16_t* Bw = (EPI == EPI_GATE2) ? ((mtile < (BB / 16)) ? B0 : B1) : B0;

  const int mrow = lane & 15;
  const int kh   = (lane >> 4) * 8;
  const int kb   = (lane >> 4) * 8;

  const bf16_t* Ap = A  + (size_t)(mtile * 16 + mrow) * lda + kh;
  const bf16_t* Bp = Bw + (size_t)(nbase + mrow) * Kdim + kb;
  const size_t bstep = (size_t)16 * Kdim;

  v8f acc[4] = {};
  v16bf a0, a1, b0[4], b1[4];

  a0 = ldA(Ap);
#pragma unroll
  for (int s = 0; s < 4; ++s) b0[s] = ldB(Bp + (size_t)s * bstep);

  const int nk = Kdim >> 5;
  int it = 1;
  for (; it + 1 < nk; it += 2) {
    const bf16_t* Ap1 = Ap + it * 32;
    const bf16_t* Bp1 = Bp + it * 32;
    a1 = ldA(Ap1);
#pragma unroll
    for (int s = 0; s < 4; ++s) b1[s] = ldB(Bp1 + (size_t)s * bstep);
    __builtin_prefetch((const void*)(Bp1 + 64), 0, 1);
#pragma unroll
    for (int s = 0; s < 4; ++s) acc[s] = wmma_bf16(a0, b0[s], acc[s]);

    const bf16_t* Ap2 = Ap + (it + 1) * 32;
    const bf16_t* Bp2 = Bp + (it + 1) * 32;
    a0 = ldA(Ap2);
#pragma unroll
    for (int s = 0; s < 4; ++s) b0[s] = ldB(Bp2 + (size_t)s * bstep);
#pragma unroll
    for (int s = 0; s < 4; ++s) acc[s] = wmma_bf16(a1, b1[s], acc[s]);
  }
  if (it < nk) {  // one trailing chunk (nk even)
    const bf16_t* Ap1 = Ap + it * 32;
    const bf16_t* Bp1 = Bp + it * 32;
    a1 = ldA(Ap1);
#pragma unroll
    for (int s = 0; s < 4; ++s) b1[s] = ldB(Bp1 + (size_t)s * bstep);
#pragma unroll
    for (int s = 0; s < 4; ++s) acc[s] = wmma_bf16(a0, b0[s], acc[s]);
#pragma unroll
    for (int s = 0; s < 4; ++s) acc[s] = wmma_bf16(a1, b1[s], acc[s]);
  } else {
#pragma unroll
    for (int s = 0; s < 4; ++s) acc[s] = wmma_bf16(a0, b0[s], acc[s]);
  }

  const int mg0   = mtile * 16 + (lane >> 4) * 8;
  const int nlane = lane & 15;
#pragma unroll
  for (int s = 0; s < 4; ++s) {
    const int n = nbase + s * 16 + nlane;
#pragma unroll
    for (int r = 0; r < 8; ++r) {
      const int m = mg0 + r;
      float v = acc[s][r];
      if (EPI == EPI_GATES) {
        outF[(size_t)m * Npad + n] = v + bias0[n] + bias1[n];
      } else if (EPI == EPI_LOGITS) {
        // contiguous per-step slice lbuf[t][b][n]
        if (n < VV)
          outF[((size_t)tcol * BB + m) * VPAD + n] = v + bias0[n];
      } else if (EPI == EPI_QTANH) {
        outB[(size_t)m * Npad + n] = f2bf(tanhf(v + bias0[n]));
      } else {  // EPI_GATE2
        if (m < BB) outB[(size_t)m * (2 * DD) + DD + n] = f2bf(tanhf(v + bias0[n]));
        else        outF[(size_t)(m - BB) * DD + n] = tanhf(v + bias1[n]);
      }
    }
  }
}

// ---------------------------------------------------------------------------
__global__ __launch_bounds__(256) void lstm_elem(const float* __restrict__ g,
                                                 float* __restrict__ c_state,
                                                 bf16_t* __restrict__ hc) {
  int i = blockIdx.x * 256 + threadIdx.x;  // < BB*DD
  int b = i / DD, d = i % DD;
  const float* gr = g + (size_t)b * 4 * DD;
  float ig = sigm(gr[d]);
  float fg = sigm(gr[DD + d]);
  float gg = tanhf(gr[2 * DD + d]);
  float og = sigm(gr[3 * DD + d]);
  float c  = fg * c_state[i] + ig * gg;
  float h  = og * tanhf(c);
  c_state[i] = c;
  hc[i] = f2bf(h);
  hc[(size_t)BB * DD + i] = f2bf(c);
}

// ---------------------------------------------------------------------------
// Argmax over contiguous logits slice lbuf[t][b][:] + embedding gather
// ---------------------------------------------------------------------------
__global__ __launch_bounds__(256) void argmax_embed(const float* __restrict__ lbuf, int tcol,
                                                    const float* __restrict__ embed,
                                                    bf16_t* __restrict__ xh) {
  __shared__ float sv[256];
  __shared__ int   si[256];
  __shared__ int   sidx;
  int b = blockIdx.x, tid = threadIdx.x;
  const float* row = lbuf + ((size_t)tcol * BB + b) * VPAD;
  float best = -3.4e38f;
  int bi = 0;
  for (int n = tid; n < VV; n += 256) {
    float v = row[n];
    if (v > best) { best = v; bi = n; }
  }
  sv[tid] = best; si[tid] = bi; __syncthreads();
  for (int o = 128; o > 0; o >>= 1) {
    if (tid < o) {
      if (sv[tid + o] > sv[tid] || (sv[tid + o] == sv[tid] && si[tid + o] < si[tid])) {
        sv[tid] = sv[tid + o]; si[tid] = si[tid + o];
      }
    }
    __syncthreads();
  }
  if (tid == 0) sidx = si[0];
  __syncthreads();
  const float* erow = embed + (size_t)sidx * DD;
  for (int d = tid; d < DD; d += 256) xh[(size_t)b * 2 * DD + d] = f2bf(erow[d]);
}

// ---------------------------------------------------------------------------
// Attention scores + softmax (wave-per-k, shfl_xor cross-lane reduction)
// ---------------------------------------------------------------------------
__global__ __launch_bounds__(256) void attn_scores(const bf16_t* __restrict__ q,
                                                   const bf16_t* __restrict__ kpre,
                                                   float* __restrict__ wout) {
  __shared__ float sq[DD];
  __shared__ float sk[64];
  int r = blockIdx.x, tid = threadIdx.x, b = r & (BB - 1);
  int lane = tid & 31, wv = tid >> 5;
  for (int d = tid; d < DD; d += 256) sq[d] = bf2f(q[(size_t)r * DD + d]);
  __syncthreads();
  for (int kk = wv; kk < KK; kk += 8) {
    const bf16_t* kr = kpre + ((size_t)b * KK + kk) * DD;
    float p = 0.f;
    for (int d = lane; d < DD; d += 32) p += sq[d] * bf2f(kr[d]);
#pragma unroll
    for (int o = 16; o > 0; o >>= 1) p += __shfl_xor(p, o, 32);
    if (lane == 0) sk[kk] = p * (1.0f / 7.0f);
  }
  __syncthreads();
  if (tid == 0) {
    float mx = sk[0];
    for (int k = 1; k < KK; ++k) mx = fmaxf(mx, sk[k]);
    float s = 0.f;
    for (int k = 0; k < KK; ++k) { float e = expf(sk[k] - mx); sk[k] = e; s += e; }
    float inv = 1.0f / s;
    for (int k = 0; k < KK; ++k) wout[(size_t)r * KK + k] = sk[k] * inv;
  }
}

// ---------------------------------------------------------------------------
__global__ __launch_bounds__(256) void attn_ctx(const float* __restrict__ wattn,
                                                const bf16_t* __restrict__ vpre,
                                                const bf16_t* __restrict__ hc,
                                                bf16_t* __restrict__ A2) {
  int r = blockIdx.x, b = r & (BB - 1);
  int d = blockIdx.y * 256 + threadIdx.x;
  __shared__ float w[KK];
  if (threadIdx.x < KK) w[threadIdx.x] = wattn[(size_t)r * KK + threadIdx.x];
  __syncthreads();
  float acc = 0.f;
  for (int kk = 0; kk < KK; ++kk) acc += w[kk] * bf2f(vpre[((size_t)b * KK + kk) * DD + d]);
  A2[(size_t)r * 2 * DD + d]      = f2bf(acc);
  A2[(size_t)r * 2 * DD + DD + d] = hc[(size_t)r * DD + d];
}

// ---------------------------------------------------------------------------
// One-time: k = tanh(ch@Wk+bk), v = tanh(ch@Wv+bv) (step-invariant)
// ---------------------------------------------------------------------------
__global__ __launch_bounds__(256) void kv_pre(const float* __restrict__ ch,
                                              const float* __restrict__ Wk, const float* __restrict__ bk,
                                              const float* __restrict__ Wv, const float* __restrict__ bv,
                                              bf16_t* __restrict__ kpre, bf16_t* __restrict__ vpre) {
  __shared__ float sK[KK * KK], sV[KK * KK], sbk[KK], sbv[KK];
  int tid = threadIdx.x;
  for (int i = tid; i < KK * KK; i += 256) { sK[i] = Wk[i]; sV[i] = Wv[i]; }
  if (tid < KK) { sbk[tid] = bk[tid]; sbv[tid] = bv[tid]; }
  __syncthreads();
  int b = blockIdx.x;
  int d = blockIdx.y * 256 + tid;
  const float* cr = ch + ((size_t)b * DD + d) * KK;
  float cv[KK];
#pragma unroll
  for (int j = 0; j < KK; ++j) cv[j] = cr[j];
  for (int kk = 0; kk < KK; ++kk) {
    float aK = sbk[kk], aV = sbv[kk];
#pragma unroll
    for (int j = 0; j < KK; ++j) { aK += cv[j] * sK[j * KK + kk]; aV += cv[j] * sV[j * KK + kk]; }
    size_t o = ((size_t)b * KK + kk) * DD + d;
    kpre[o] = f2bf(tanhf(aK));
    vpre[o] = f2bf(tanhf(aV));
  }
}

// ---------------------------------------------------------------------------
// Weight convert+transpose (fp32 [K,N] -> bf16 [Npad][K]); optional K-stacked 2nd src
// ---------------------------------------------------------------------------
__global__ void conv_tr(const float* __restrict__ in0, const float* __restrict__ in1,
                        bf16_t* __restrict__ outT, int N, int K0, int Ktot, long total) {
  long i = (long)blockIdx.x * 256 + threadIdx.x;
  if (i >= total) return;
  int n = (int)(i / Ktot), k = (int)(i % Ktot);
  float v = 0.f;
  if (n < N) v = (k < K0) ? in0[(size_t)k * N + n] : in1[(size_t)(k - K0) * N + n];
  outT[i] = f2bf(v);
}

// row0[n] = embed[SOS] @ Wp + bp (shared by all batch rows of slice 0)
__global__ void x0_row(const float* __restrict__ embed, const float* __restrict__ Wp,
                       const float* __restrict__ bp, float* __restrict__ row0) {
  int n = blockIdx.x * 256 + threadIdx.x;
  if (n >= VPAD) return;
  float acc = -3.0e38f;
  if (n < VV) {
    acc = bp[n];
    for (int d = 0; d < DD; ++d) acc += embed[d] * Wp[(size_t)d * VV + n];
  }
  row0[n] = acc;
}

__global__ void x0_bcast(const float* __restrict__ row0, float* __restrict__ lbuf) {
  long i = (long)blockIdx.x * 256 + threadIdx.x;  // < BB*VPAD
  if (i >= (long)BB * VPAD) return;
  lbuf[i] = row0[i % VPAD];
}

// Final emit: lbuf[t][b][n] -> out[b][n][t]. Coalesced reads (lane-adjacent n),
// each thread writes 40 contiguous floats (full 64B-line coverage).
__global__ void emit_out(const float* __restrict__ lbuf, float* __restrict__ out) {
  long i = (long)blockIdx.x * 256 + threadIdx.x;  // < BB*VV
  if (i >= (long)BB * VV) return;
  int b = (int)(i / VV), n = (int)(i % VV);
  float* dst = out + ((size_t)b * VV + n) * TT;
  for (int t = 0; t < TT; ++t) dst[t] = lbuf[((size_t)t * BB + b) * VPAD + n];
}

__global__ void init_state(const float* __restrict__ embed, const float* __restrict__ pooled,
                           bf16_t* __restrict__ xh, float* __restrict__ c_state) {
  int i = blockIdx.x * 256 + threadIdx.x;  // < BB*DD
  int b = i / DD, d = i % DD;
  xh[(size_t)b * 2 * DD + d]      = f2bf(embed[d]);   // embed[SOS=0]
  xh[(size_t)b * 2 * DD + DD + d] = f2bf(pooled[i]);  // h0 = pooled
  c_state[i] = pooled[i];                             // c0 = pooled
}

// ---------------------------------------------------------------------------
extern "C" void kernel_launch(void* const* d_in, const int* in_sizes, int n_in,
                              void* d_out, int out_size, void* d_ws, size_t ws_size,
                              hipStream_t stream) {
  (void)in_sizes; (void)n_in; (void)out_size; (void)ws_size;
  const float* img    = (const float*)d_in[0];
  const float* pooled = (const float*)d_in[1];
  const float* W_ih   = (const float*)d_in[2];
  const float* W_hh   = (const float*)d_in[3];
  const float* b_ih   = (const float*)d_in[4];
  const float* b_hh   = (const float*)d_in[5];
  const float* Wq     = (const float*)d_in[6];
  const float* bq     = (const float*)d_in[7];
  const float* Wk     = (const float*)d_in[8];
  const float* bk     = (const float*)d_in[9];
  const float* Wv     = (const float*)d_in[10];
  const float* bv     = (const float*)d_in[11];
  const float* Wh     = (const float*)d_in[12];
  const float* bh     = (const float*)d_in[13];
  const float* Wc     = (const float*)d_in[14];
  const float* bc     = (const float*)d_in[15];
  const float* Wp     = (const float*)d_in[16];
  const float* bp     = (const float*)d_in[17];
  const float* embed  = (const float*)d_in[18];
  float* out = (float*)d_out;

  char* p = (char*)d_ws;
  auto take = [&](size_t bytes) -> char* {
    char* r = p; p += (bytes + 255) & ~(size_t)255; return r;
  };
  float*  lbuf = (float*)take((size_t)TT * BB * VPAD * 4);    // all-step logits (contig)
  bf16_t* WgT  = (bf16_t*)take((size_t)4 * DD * 2 * DD * 2);  // [5120][2560]
  bf16_t* WpT  = (bf16_t*)take((size_t)VPAD * DD * 2);        // [10240][1280]
  bf16_t* WqT  = (bf16_t*)take((size_t)DD * DD * 2);
  bf16_t* WhT  = (bf16_t*)take((size_t)DD * 2 * DD * 2);
  bf16_t* WcT  = (bf16_t*)take((size_t)DD * 2 * DD * 2);
  bf16_t* kpre = (bf16_t*)take((size_t)BB * KK * DD * 2);
  bf16_t* vpre = (bf16_t*)take((size_t)BB * KK * DD * 2);
  bf16_t* xh   = (bf16_t*)take((size_t)BB * 2 * DD * 2);
  float*  gbuf = (float*)take((size_t)BB * 4 * DD * 4);
  float*  c_st = (float*)take((size_t)BB * DD * 4);
  bf16_t* hc   = (bf16_t*)take((size_t)2 * BB * DD * 2);
  bf16_t* qbuf = (bf16_t*)take((size_t)2 * BB * DD * 2);
  float*  watt = (float*)take((size_t)2 * BB * KK * 4);
  bf16_t* A2   = (bf16_t*)take((size_t)2 * BB * 2 * DD * 2);
  float*  row0 = (float*)take((size_t)VPAD * 4);

  // ---- one-time setup ----
  { long tot = (long)4 * DD * 2 * DD;
    conv_tr<<<(tot + 255) / 256, 256, 0, stream>>>(W_ih, W_hh, WgT, 4 * DD, DD, 2 * DD, tot); }
  { long tot = (long)VPAD * DD;
    conv_tr<<<(tot + 255) / 256, 256, 0, stream>>>(Wp, Wp, WpT, VV, DD, DD, tot); }
  { long tot = (long)DD * DD;
    conv_tr<<<(tot + 255) / 256, 256, 0, stream>>>(Wq, Wq, WqT, DD, DD, DD, tot); }
  { long tot = (long)DD * 2 * DD;
    conv_tr<<<(tot + 255) / 256, 256, 0, stream>>>(Wh, Wh, WhT, DD, 2 * DD, 2 * DD, tot); }
  { long tot = (long)DD * 2 * DD;
    conv_tr<<<(tot + 255) / 256, 256, 0, stream>>>(Wc, Wc, WcT, DD, 2 * DD, 2 * DD, tot); }
  kv_pre<<<dim3(BB, DD / 256), 256, 0, stream>>>(img, Wk, bk, Wv, bv, kpre, vpre);
  x0_row<<<(VPAD + 255) / 256, 256, 0, stream>>>(embed, Wp, bp, row0);
  x0_bcast<<<((long)BB * VPAD + 255) / 256, 256, 0, stream>>>(row0, lbuf);
  init_state<<<(BB * DD) / 256, 256, 0, stream>>>(embed, pooled, xh, c_st);

  // ---- recurrent decode: 39 steps ----
  for (int t = 1; t < TT; ++t) {
    gemm_wmma<EPI_GATES><<<dim3(20, 8), 128, 0, stream>>>(
        xh, 2 * DD, WgT, nullptr, 2 * DD, 4 * DD, gbuf, nullptr, b_ih, b_hh, 0);
    lstm_elem<<<(BB * DD) / 256, 256, 0, stream>>>(gbuf, c_st, hc);
    gemm_wmma<EPI_LOGITS><<<dim3(40, 8), 128, 0, stream>>>(
        hc, DD, WpT, nullptr, DD, VPAD, lbuf, nullptr, bp, nullptr, t);
    argmax_embed<<<BB, 256, 0, stream>>>(lbuf, t, embed, xh);
    gemm_wmma<EPI_QTANH><<<dim3(5, 16), 128, 0, stream>>>(
        hc, DD, WqT, nullptr, DD, DD, nullptr, qbuf, bq, nullptr, 0);
    attn_scores<<<2 * BB, 256, 0, stream>>>(qbuf, kpre, watt);
    attn_ctx<<<dim3(2 * BB, DD / 256), 256, 0, stream>>>(watt, vpre, hc, A2);
    gemm_wmma<EPI_GATE2><<<dim3(5, 16), 128, 0, stream>>>(
        A2, 2 * DD, WhT, WcT, 2 * DD, DD, c_st, xh, bh, bc, 0);
  }

  // ---- final transpose into [B, V, T] ----
  emit_out<<<((long)BB * VV + 255) / 256, 256, 0, stream>>>(lbuf, out);
}
